// GCNN_417
// MI455X (gfx1250) — matrix operand order repaired and checked by reading, not compiled, not run on hardware
//
#include <hip/hip_runtime.h>
#include <hip/hip_bf16.h>
#include <math.h>

// ---- CDNA5 WMMA vector types -------------------------------------------------
typedef __attribute__((ext_vector_type(16))) __bf16 v16bf;
typedef __attribute__((ext_vector_type(8)))  float  v8f;
typedef __attribute__((ext_vector_type(4)))  unsigned int v4u;

union V16BF { v16bf v; unsigned short u[16]; v4u q[2]; };

// Hardware bf16 convert (v_cvt_*_bf16_f32) instead of software RNE rounding.
__device__ __forceinline__ unsigned short f2bf(float f) {
    __bf16 h = (__bf16)f;
    return __builtin_bit_cast(unsigned short, h);
}

// Map K index k (0..31) within a 32-wide chunk to the A-operand slot:
//   half (lane>=16?), e (entry 0..15 inside the lane's 16 bf16)
// A layout (16-bit A 16x32): lane<16 K={0..7,16..23}, lane>=16 K={8..15,24..31}, 2/VGPR.
__device__ __forceinline__ void a_slot(int k, int& half, int& e) {
    int p = k & 1, kk = k & ~1;
    int h, v;
    if (kk < 16) { h = (kk >= 8) ? 1 : 0; v = (kk - 8 * h) >> 1; }
    else { int z = kk - 16; h = (z >= 8) ? 1 : 0; v = 4 + ((z - 8 * h) >> 1); }
    half = h; e = 2 * v + p;
}

// B layout (16-bit B 32x16): lane<16 K={0..7,16..23}, lane>=16 K={8..15,24..31}, 2/VGPR.
__device__ __forceinline__ void b_slot(int k, int& half, int& e) {
    half = (k >> 3) & 1;
    e = (k & 7) + ((k >> 4) << 3);
}

// ---- Sizes (fixed by the reference) -----------------------------------------
#define NB   4
#define NN   128
#define NF   32
#define KTOT 544     // 16*32 (T flattened) + 32 (x row)  == 17 * 32

// =============================================================================
// Kernel 1: A-swizzled Maug.
//   Logical Maug[b,i,j,s]: s<8 = a*e8, s==8 = a, s>8 = 0  (bf16)
//   Stored per (b,i) in stage-1 A-operand order: [jc(4)][lane(32)][16] bf16.
// =============================================================================
__global__ __launch_bounds__(256) void build_maug(
        const float* __restrict__ e, const float* __restrict__ a,
        const float* __restrict__ emb_w, unsigned short* __restrict__ M) {
    int idx = blockIdx.x * blockDim.x + threadIdx.x;   // < 4*128*128
    if (idx >= NB * NN * NN) return;
    int j = idx & (NN - 1);
    int bi = idx >> 7;                                  // b*128 + i
    const float* er = e + (size_t)idx * 16;
    float ev[16];
#pragma unroll
    for (int t = 0; t < 16; ++t) ev[t] = er[t];
    float aa = a[idx];
    unsigned short outv[16];
#pragma unroll
    for (int s = 0; s < 8; ++s) {
        float acc = 0.f;
#pragma unroll
        for (int t = 0; t < 16; ++t) acc += ev[t] * emb_w[t * 8 + s];
        outv[s] = f2bf(acc * aa);
    }
    outv[8] = f2bf(aa);
#pragma unroll
    for (int s = 9; s < 16; ++s) outv[s] = 0;

    int jc = j >> 5, k = j & 31;
    int half, en;
    a_slot(k, half, en);
    unsigned short* base = M + (((size_t)bi * 4 + jc) * 32) * 16;
#pragma unroll
    for (int s = 0; s < 16; ++s)
        base[(size_t)(s + 16 * half) * 16 + en] = outv[s];
}

// =============================================================================
// Kernel 2: B-swizzled combined second-stage weight.
//   Logical W2[c][o], c in [0,544):
//     c<512: (s=c/32,f=c%32) -> s<8: kn_w[s, o*32+f]; s==8: kn_b[o*32+f]; else 0
//     c>=512: root[c-512][o]
//   Stored in stage-2 B-operand order: [tile(2)][kc(17)][lane(32)][16] bf16.
// =============================================================================
__global__ __launch_bounds__(256) void build_w2(
        const float* __restrict__ kn_w, const float* __restrict__ kn_b,
        const float* __restrict__ root, unsigned short* __restrict__ W2) {
    int idx = blockIdx.x * blockDim.x + threadIdx.x;   // < 544*32
    if (idx >= KTOT * NF) return;
    int c = idx >> 5, o = idx & 31;
    float v;
    if (c < 512) {
        int s = c >> 5, f = c & 31;
        v = (s < 8) ? kn_w[s * 1024 + o * 32 + f] : (s == 8 ? kn_b[o * 32 + f] : 0.f);
    } else {
        v = root[(c - 512) * NF + o];
    }
    int kc = c >> 5, k = c & 31;
    int half, en;
    b_slot(k, half, en);
    int tile = o >> 4, n = o & 15;
    W2[(((size_t)tile * 17 + kc) * 32 + (n + 16 * half)) * 16 + en] = f2bf(v);
}

// =============================================================================
// Kernel 3: one ECC layer (factored), fused tanh + batchnorm.
// Grid: 32 blocks = (b) x (iTile), 256 threads = 8 waves.
// All WMMA operands are pre-swizzled -> every operand load is 2x B128.
// =============================================================================
__global__ __launch_bounds__(256) void ecc_layer(
        const unsigned short* __restrict__ MaugS, const unsigned short* __restrict__ W2S,
        const float* __restrict__ xin, float* __restrict__ hout,
        const float* __restrict__ bias,
        const float* __restrict__ bng, const float* __restrict__ bnb,
        const float* __restrict__ bnm, const float* __restrict__ bnv) {
    __shared__ unsigned short xs[4 * 2 * 32 * 16];   // stage-1 B operands, 8 KB
    __shared__ unsigned short xaS[32 * 16];          // stage-2 A tail (tile's own x), 1 KB
    __shared__ unsigned short TallS[16 * 32 * 16];   // stage-2 A chunks kc=0..15, 16 KB

    int b = blockIdx.x >> 3;
    int iTile = blockIdx.x & 7;
    int i0 = iTile * 16;
    int tid = threadIdx.x;
    const float* xb = xin + (size_t)b * NN * NF;

    // ---- stage x into LDS, already swizzled ----
    // xs[jc][tile][lane][e] = x[b][jc*32 + k][n + 16*tile],  k from B-slot(lane,e)
    for (int d = tid; d < 4096; d += 256) {
        int en = d & 15, lane = (d >> 4) & 31, tile = (d >> 9) & 1, jc = d >> 10;
        int half = lane >> 4, n = lane & 15;
        int k = (en < 8) ? 8 * half + en : 8 + 8 * half + en;
        xs[d] = f2bf(xb[(size_t)(jc * 32 + k) * NF + (n + 16 * tile)]);
    }
    // xaS[lane][e] = x[b][i0 + m][k],  m = lane&15, k from A-slot(lane,e)
    for (int d = tid; d < 512; d += 256) {
        int en = d & 15, lane = d >> 4;
        int half = lane >> 4, m = lane & 15;
        int v = en >> 1, p = en & 1;
        int k = (v < 4 ? 2 * v : 16 + 2 * (v - 4)) + 8 * half + p;
        xaS[d] = f2bf(xb[(size_t)(i0 + m) * NF + k]);
    }
    __syncthreads();

    int wave = tid >> 5, lane = tid & 31;
    int n = lane & 15, half = lane >> 4;

    // ---- stage 1: per wave, rows il = wave*2 + {0,1} ----
    //   T[16s x 32f] = Maug_i^T (16x128) @ x (128x32)
    for (int r = 0; r < 2; ++r) {
        int il = wave * 2 + r;
        const unsigned short* Mbase =
            MaugS + ((size_t)(b * NN + i0 + il) * 4 * 32) * 16;
        if (r == 0)   // prefetch next row's A slab (global_prefetch_b8)
            __builtin_prefetch(MaugS + ((size_t)(b * NN + i0 + il + 1) * 4 * 32) * 16, 0, 1);
        v8f clo = {}, chi = {};
#pragma unroll
        for (int jc = 0; jc < 4; ++jc) {
            V16BF A, Blo, Bhi;
            const v4u* ap = (const v4u*)(Mbase + ((size_t)jc * 32 + lane) * 16);
            A.q[0] = ap[0]; A.q[1] = ap[1];
            const v4u* bl = (const v4u*)(xs + ((size_t)(jc * 2 + 0) * 32 + lane) * 16);
            Blo.q[0] = bl[0]; Blo.q[1] = bl[1];
            const v4u* bh = (const v4u*)(xs + ((size_t)(jc * 2 + 1) * 32 + lane) * 16);
            Bhi.q[0] = bh[0]; Bhi.q[1] = bh[1];
            clo = __builtin_amdgcn_wmma_f32_16x16x32_bf16(false, A.v, false, Blo.v,
                                                          (short)0, clo, false, false);
            chi = __builtin_amdgcn_wmma_f32_16x16x32_bf16(false, A.v, false, Bhi.v,
                                                          (short)0, chi, false, false);
        }
        // scatter T into stage-2 A-operand order:
        //   chunk kc == s (since c = s*32+f), row M = il, K-within-chunk = f
#pragma unroll
        for (int v = 0; v < 8; ++v) {
            int s = v + half * 8;
            int h2, e2;
            a_slot(n, h2, e2);                  // f = n
            TallS[((size_t)s * 32 + (il + 16 * h2)) * 16 + e2] = f2bf(clo[v]);
            a_slot(n + 16, h2, e2);             // f = n + 16
            TallS[((size_t)s * 32 + (il + 16 * h2)) * 16 + e2] = f2bf(chi[v]);
        }
    }
    __syncthreads();

    // ---- stage 2: waves 0,1 -> Out[16i x 32o] = [T|x](16x544) @ W2(544x32) ----
    if (wave < 2) {
        const unsigned short* Wbase = W2S + ((size_t)wave * 17 * 32) * 16;
        v8f acc = {};
#pragma unroll 1
        for (int kc = 0; kc < 17; ++kc) {
            V16BF A, Bm;
            const v4u* ap = (kc < 16)
                ? (const v4u*)(TallS + ((size_t)kc * 32 + lane) * 16)
                : (const v4u*)(xaS + (size_t)lane * 16);
            A.q[0] = ap[0]; A.q[1] = ap[1];
            const v4u* bp = (const v4u*)(Wbase + ((size_t)kc * 32 + lane) * 16);
            Bm.q[0] = bp[0]; Bm.q[1] = bp[1];
            acc = __builtin_amdgcn_wmma_f32_16x16x32_bf16(false, A.v, false, Bm.v,
                                                          (short)0, acc, false, false);
        }
        // epilogue: + bias, tanh, batchnorm (eps = 1e-3), store f32
        int o = n + wave * 16;
        float bo = bias[o], go = bng[o], bbo = bnb[o], mo = bnm[o];
        float inv = rsqrtf(bnv[o] + 1e-3f);
#pragma unroll
        for (int v = 0; v < 8; ++v) {
            int il = v + half * 8;
            float val = tanhf(acc[v] + bo);
            val = (val - mo) * inv * go + bbo;
            hout[((size_t)(b * NN + i0 + il)) * NF + o] = val;
        }
    }
}

// =============================================================================
// Kernel 4: GlobalAttnSumPool + Dense(1024, LeakyReLU) + BN + MLE head.
// =============================================================================
__global__ __launch_bounds__(1024) void readout(
        const float* __restrict__ h, const float* __restrict__ attn_k,
        const float* __restrict__ dense_w, const float* __restrict__ dense_b,
        const float* __restrict__ bnf_g, const float* __restrict__ bnf_b,
        const float* __restrict__ bnf_m, const float* __restrict__ bnf_v,
        const float* __restrict__ mle_w, const float* __restrict__ mle_b,
        const float* __restrict__ sigma, float* __restrict__ out) {
    __shared__ float lg[NB * NN];
    __shared__ float aw[NB * NN];
    __shared__ float gs[NB * NF];
    __shared__ float g2[NB * 1024];
    __shared__ float rsum[NB];
    int tid = threadIdx.x;

    if (tid < NB * NN) {
        float acc = 0.f;
#pragma unroll
        for (int f = 0; f < NF; ++f) acc += h[(size_t)tid * NF + f] * attn_k[f];
        lg[tid] = acc;
    }
    __syncthreads();
    if (tid < NB) {
        float mx = -1e30f;
        for (int nn = 0; nn < NN; ++nn) mx = fmaxf(mx, lg[tid * NN + nn]);
        float s = 0.f;
        for (int nn = 0; nn < NN; ++nn) { float e = expf(lg[tid * NN + nn] - mx); aw[tid * NN + nn] = e; s += e; }
        rsum[tid] = 1.f / s;
    }
    __syncthreads();
    if (tid < NB * NN) aw[tid] *= rsum[tid >> 7];
    __syncthreads();
    if (tid < NB * NF) {
        int b = tid >> 5, f = tid & 31;
        float acc = 0.f;
        for (int nn = 0; nn < NN; ++nn) acc += aw[b * NN + nn] * h[(size_t)(b * NN + nn) * NF + f];
        gs[tid] = acc;
    }
    __syncthreads();
    {
        int o = tid;
        float inv = rsqrtf(bnf_v[o] + 1e-3f);
        float go = bnf_g[o], bo = bnf_b[o], mo = bnf_m[o], db = dense_b[o];
#pragma unroll 1
        for (int b = 0; b < NB; ++b) {
            float acc = db;
#pragma unroll
            for (int f = 0; f < NF; ++f) acc += gs[b * NF + f] * dense_w[f * 1024 + o];
            acc = (acc > 0.f) ? acc : 0.01f * acc;
            g2[b * 1024 + o] = (acc - mo) * inv * go + bo;
        }
    }
    __syncthreads();
    if (tid < NB) {
        float acc = mle_b[0];
        for (int k = 0; k < 1024; ++k) acc += g2[tid * 1024 + k] * mle_w[k];
        out[tid] = acc;
    }
    if (tid == NB) out[NB] = sigma[0];
}

// =============================================================================
extern "C" void kernel_launch(void* const* d_in, const int* in_sizes, int n_in,
                              void* d_out, int out_size, void* d_ws, size_t ws_size,
                              hipStream_t stream) {
    const float* x      = (const float*)d_in[0];
    const float* a      = (const float*)d_in[1];
    const float* e      = (const float*)d_in[2];
    const float* emb_w  = (const float*)d_in[3];
    const float* kn_w0  = (const float*)d_in[4];
    const float* kn_b0  = (const float*)d_in[5];
    const float* root0  = (const float*)d_in[6];
    const float* bias0  = (const float*)d_in[7];
    const float* bn_g0  = (const float*)d_in[8];
    const float* bn_b0  = (const float*)d_in[9];
    const float* bn_m0  = (const float*)d_in[10];
    const float* bn_v0  = (const float*)d_in[11];
    const float* kn_w1  = (const float*)d_in[12];
    const float* kn_b1  = (const float*)d_in[13];
    const float* root1  = (const float*)d_in[14];
    const float* bias1  = (const float*)d_in[15];
    const float* bn_g1  = (const float*)d_in[16];
    const float* bn_b1  = (const float*)d_in[17];
    const float* bn_m1  = (const float*)d_in[18];
    const float* bn_v1  = (const float*)d_in[19];
    const float* attn_k = (const float*)d_in[20];
    const float* dense_w= (const float*)d_in[21];
    const float* dense_b= (const float*)d_in[22];
    const float* bnf_g  = (const float*)d_in[23];
    const float* bnf_b  = (const float*)d_in[24];
    const float* bnf_m  = (const float*)d_in[25];
    const float* bnf_v  = (const float*)d_in[26];
    const float* mle_w  = (const float*)d_in[27];
    const float* mle_b  = (const float*)d_in[28];
    const float* sigma  = (const float*)d_in[29];

    // workspace layout (bytes)
    char* ws = (char*)d_ws;
    unsigned short* MaugS = (unsigned short*)ws;                        // 2,097,152 B
    unsigned short* W2S0  = (unsigned short*)(ws + 2097152);            //    34,816 B
    unsigned short* W2S1  = W2S0 + (size_t)KTOT * NF;                   //    34,816 B
    float* h0 = (float*)(ws + 2097152 + 2 * 34816);                     //    65,536 B
    float* h1 = h0 + (size_t)NB * NN * NF;                              //    65,536 B

    build_maug<<<(NB * NN * NN) / 256, 256, 0, stream>>>(e, a, emb_w, MaugS);
    build_w2<<<(KTOT * NF + 255) / 256, 256, 0, stream>>>(kn_w0, kn_b0, root0, W2S0);
    build_w2<<<(KTOT * NF + 255) / 256, 256, 0, stream>>>(kn_w1, kn_b1, root1, W2S1);

    ecc_layer<<<NB * (NN / 16), 256, 0, stream>>>(MaugS, W2S0, x,  h0,
                                                  bias0, bn_g0, bn_b0, bn_m0, bn_v0);
    ecc_layer<<<NB * (NN / 16), 256, 0, stream>>>(MaugS, W2S1, h0, h1,
                                                  bias1, bn_g1, bn_b1, bn_m1, bn_v1);

    readout<<<1, 1024, 0, stream>>>(h1, attn_k, dense_w, dense_b,
                                    bnf_g, bnf_b, bnf_m, bnf_v,
                                    mle_w, mle_b, sigma, (float*)d_out);
}
